// SpatialContextAttention_9758165697077
// MI455X (gfx1250) — matrix operand order repaired and checked by reading, not compiled, not run on hardware
//
#include <hip/hip_runtime.h>
#include <cstdint>

#define BATCH 4
#define CDIM  256
#define NSEQ  4096
#define QKD   32

typedef __attribute__((ext_vector_type(16))) __bf16 v16bf;
typedef __attribute__((ext_vector_type(8)))  float  v8f;

// ---------- helpers ----------

__device__ __forceinline__ uint16_t f2bf(float f) {
  union { float f; uint32_t u; } x; x.f = f;
  uint32_t r = x.u + 0x7FFFu + ((x.u >> 16) & 1u);   // round-to-nearest-even
  return (uint16_t)(r >> 16);
}

union FragU { v16bf v; uint4 q[2]; };

// A-fragment (16x32 bf16): rowPtr -> this lane's row (lane&15).
// Per ISA: lane half h holds K = k0+8h+{0..7} and K = k0+16+8h+{0..7}.
__device__ __forceinline__ v16bf load_a(const uint16_t* rowPtr, int k0, int lane) {
  const int h = (lane >> 4) << 3;
  FragU f;
  f.q[0] = *(const uint4*)(rowPtr + k0 + h);
  f.q[1] = *(const uint4*)(rowPtr + k0 + 16 + h);
  return f.v;
}

// B-fragment (32x16 bf16): colPtr -> start of this lane's column K-run region
// (column = lane&15 resolved by caller). Lane half h holds K = 8h+{0..7} and K = 16+8h+{0..7}.
__device__ __forceinline__ v16bf load_b(const uint16_t* colPtr, int lane) {
  const int h = (lane >> 4) << 3;
  FragU f;
  f.q[0] = *(const uint4*)(colPtr + h);
  f.q[1] = *(const uint4*)(colPtr + 16 + h);
  return f.v;
}

__device__ __forceinline__ v8f wmma_bf16(v16bf a, v16bf b, v8f c) {
  return __builtin_amdgcn_wmma_f32_16x16x32_bf16(false, a, false, b, (short)0, c,
                                                 false, false);
}

// ---------- kernel 0a: fp32 -> bf16 cast (weights) ----------

__global__ void cast_bf16_kernel(const float* __restrict__ in,
                                 uint16_t* __restrict__ out, int n) {
  int i = blockIdx.x * 256 + threadIdx.x;
  if (i < n) out[i] = f2bf(in[i]);
}

// ---------- kernel 0b: x (B,C,N) fp32 -> xT (B,N,C) bf16 ----------

__global__ void transpose_cast_kernel(const float* __restrict__ x,
                                      uint16_t* __restrict__ xT) {
  __shared__ float t[32][33];
  const int b  = blockIdx.z;
  const int n0 = blockIdx.x * 32;
  const int c0 = blockIdx.y * 32;
  const float* xb = x + (size_t)b * CDIM * NSEQ;
  for (int i = threadIdx.y; i < 32; i += 8)
    t[i][threadIdx.x] = xb[(size_t)(c0 + i) * NSEQ + n0 + threadIdx.x];
  __syncthreads();
  uint16_t* xTb = xT + (size_t)b * NSEQ * CDIM;
  for (int i = threadIdx.y; i < 32; i += 8)
    xTb[(size_t)(n0 + i) * CDIM + c0 + threadIdx.x] = f2bf(t[threadIdx.x][i]);
}

// ---------- kernel 1: QKV projections (WMMA) ----------
// qT,kT: (B,N,32) bf16 (q pre-scaled by 1/sqrt(32)*log2(e)); v: (B,C,N) bf16

__global__ __launch_bounds__(128, 1) void qkv_kernel(
    const uint16_t* __restrict__ xT,
    const uint16_t* __restrict__ wqb, const uint16_t* __restrict__ wkb,
    const uint16_t* __restrict__ wvb,
    const float* __restrict__ bq, const float* __restrict__ bk,
    const float* __restrict__ bv,
    uint16_t* __restrict__ qT, uint16_t* __restrict__ kT,
    uint16_t* __restrict__ vbuf) {
  const int lane  = threadIdx.x & 31;
  const int wave  = threadIdx.x >> 5;
  const int b     = blockIdx.x >> 6;           // NSEQ/64 == 64 tiles per batch
  const int ntile = blockIdx.x & 63;
  const int nbase = ntile * 64 + wave * 16;

  const uint16_t* xTb = xT + (size_t)b * NSEQ * CDIM;
  const uint16_t* arow_x = xTb + (size_t)(nbase + (lane & 15)) * CDIM;

  const float QSCALE = 0.17677669529663688f * 1.4426950408889634f; // 1/sqrt(32)*log2e
  const v8f vzero = {};

  // ---- q and k : out (n rows x 32 cols), A = xT rows, B = w rows (K-contiguous)
  for (int proj = 0; proj < 2; ++proj) {
    const uint16_t* w   = proj ? wkb : wqb;
    const float* bias   = proj ? bk  : bq;
    uint16_t* outp      = proj ? kT  : qT;
    const float sc      = proj ? 1.0f : QSCALE;
    for (int t = 0; t < 2; ++t) {
      v8f acc = vzero;
      const uint16_t* bcol = w + (size_t)(t * 16 + (lane & 15)) * CDIM;
#pragma unroll
      for (int ks = 0; ks < 8; ++ks) {
        v16bf a  = load_a(arow_x, ks * 32, lane);
        v16bf bb = load_b(bcol + ks * 32, lane);
        acc = wmma_bf16(a, bb, acc);
      }
#pragma unroll
      for (int r = 0; r < 8; ++r) {     // D: lane=col(o), vgpr=row(n)
        int row = nbase + r + ((lane >> 4) << 3);
        int o   = t * 16 + (lane & 15);
        float val = (acc[r] + bias[o]) * sc;
        outp[(size_t)b * NSEQ * QKD + (size_t)row * QKD + o] = f2bf(val);
      }
    }
  }

  // ---- v : out (o rows x n cols), A = wv rows, B = xT rows of this wave's n
  const uint16_t* bcol_x = arow_x;   // column n = nbase + lane&15, K-contiguous c
  for (int t = 0; t < 16; ++t) {
    v8f acc = vzero;
    const uint16_t* arow_w = wvb + (size_t)(t * 16 + (lane & 15)) * CDIM;
#pragma unroll
    for (int ks = 0; ks < 8; ++ks) {
      v16bf a  = load_a(arow_w, ks * 32, lane);
      v16bf bb = load_b(bcol_x + ks * 32, lane);
      acc = wmma_bf16(a, bb, acc);
    }
#pragma unroll
    for (int r = 0; r < 8; ++r) {       // D: lane=col(n), vgpr=row(o)
      int o = t * 16 + r + ((lane >> 4) << 3);
      int n = nbase + (lane & 15);
      vbuf[(size_t)b * CDIM * NSEQ + (size_t)o * NSEQ + n] = f2bf(acc[r] + bv[o]);
    }
  }
}

// ---------- kernel 2: flash attention (WMMA, online softmax, KB=64) ----------
// attT: (B,N,C) bf16 = softmax(q k^T / sqrt(d)) v^T

__global__ __launch_bounds__(128, 1) void attn_kernel(
    const uint16_t* __restrict__ qT, const uint16_t* __restrict__ kT,
    const uint16_t* __restrict__ vbuf, uint16_t* __restrict__ attT) {
  __shared__ __align__(16) uint16_t lds_p[4][16][64];   // per-wave exp-score tile

  const int lane  = threadIdx.x & 31;
  const int wave  = threadIdx.x >> 5;
  const int b     = blockIdx.x >> 6;
  const int ntile = blockIdx.x & 63;
  const int nbase = ntile * 64 + wave * 16;

  const uint16_t* qTb = qT   + (size_t)b * NSEQ * QKD;
  const uint16_t* kTb = kT   + (size_t)b * NSEQ * QKD;
  const uint16_t* vb  = vbuf + (size_t)b * CDIM * NSEQ;

  // Q fragment is resident for the whole key loop (rows = this wave's 16 queries)
  const v16bf a_q = load_a(qTb + (size_t)(nbase + (lane & 15)) * QKD, 0, lane);

  const v8f vzero = {};
  v8f acc[16];
#pragma unroll
  for (int t = 0; t < 16; ++t) acc[t] = vzero;

  float mrun[8], lrun[8];
#pragma unroll
  for (int r = 0; r < 8; ++r) { mrun[r] = -1e30f; lrun[r] = 0.0f; }

  for (int mb = 0; mb < NSEQ; mb += 64) {
    // prefetch next key block (K and a slice of V) into L2 (global_prefetch_b8)
    if (mb + 64 < NSEQ) {
      __builtin_prefetch(kTb + (size_t)(mb + 64 + lane) * QKD, 0, 1);
      __builtin_prefetch(vb + (size_t)(lane * 8) * NSEQ + mb + 64, 0, 1);
    }

    // scores for 64 keys: four 16x16 WMMA tiles (K-dim = head dim = 32)
    v8f s[4];
#pragma unroll
    for (int j = 0; j < 4; ++j) {
      v16bf bk = load_b(kTb + (size_t)(mb + j * 16 + (lane & 15)) * QKD, lane);
      s[j] = wmma_bf16(a_q, bk, vzero);
    }

    // online softmax; C/D layout: lane = key column, vgpr r = row (r + 8*(lane/16))
    float alpha[8];
#pragma unroll
    for (int r = 0; r < 8; ++r) {
      float mx = fmaxf(fmaxf(s[0][r], s[1][r]), fmaxf(s[2][r], s[3][r]));
      mx = fmaxf(mx, __shfl_xor(mx, 1));
      mx = fmaxf(mx, __shfl_xor(mx, 2));
      mx = fmaxf(mx, __shfl_xor(mx, 4));
      mx = fmaxf(mx, __shfl_xor(mx, 8));          // row max within 16-lane half
      float mnew = fmaxf(mrun[r], mx);
      alpha[r] = __builtin_exp2f(mrun[r] - mnew);
      float p0 = __builtin_exp2f(s[0][r] - mnew);
      float p1 = __builtin_exp2f(s[1][r] - mnew);
      float p2 = __builtin_exp2f(s[2][r] - mnew);
      float p3 = __builtin_exp2f(s[3][r] - mnew);
      float rs = (p0 + p1) + (p2 + p3);
      rs += __shfl_xor(rs, 1);
      rs += __shfl_xor(rs, 2);
      rs += __shfl_xor(rs, 4);
      rs += __shfl_xor(rs, 8);
      lrun[r] = lrun[r] * alpha[r] + rs;
      mrun[r] = mnew;
      int row = r + ((lane >> 4) << 3);
      int col = lane & 15;
      lds_p[wave][row][col]      = f2bf(p0);
      lds_p[wave][row][col + 16] = f2bf(p1);
      lds_p[wave][row][col + 32] = f2bf(p2);
      lds_p[wave][row][col + 48] = f2bf(p3);
    }

    // rescale running accumulators by alpha (per-row, layout-matched)
#pragma unroll
    for (int t = 0; t < 16; ++t)
#pragma unroll
      for (int r = 0; r < 8; ++r) acc[t][r] *= alpha[r];

    __syncthreads();   // order LDS p-tile writes before fragment reload

    // reload P in A-fragment layout (keys 0..31 and 32..63), then P·V WMMAs
    v16bf a_p0 = load_a(&lds_p[wave][lane & 15][0], 0, lane);
    v16bf a_p1 = load_a(&lds_p[wave][lane & 15][0], 32, lane);
#pragma unroll
    for (int t = 0; t < 16; ++t) {
      const uint16_t* vcol = vb + (size_t)(t * 16 + (lane & 15)) * NSEQ + mb;
      v16bf bv0 = load_b(vcol, lane);
      v16bf bv1 = load_b(vcol + 32, lane);
      acc[t] = wmma_bf16(a_p0, bv0, acc[t]);
      acc[t] = wmma_bf16(a_p1, bv1, acc[t]);
    }
    __syncthreads();
  }

#pragma unroll
  for (int r = 0; r < 8; ++r) lrun[r] = 1.0f / lrun[r];

  uint16_t* attb = attT + (size_t)b * NSEQ * CDIM;
#pragma unroll
  for (int t = 0; t < 16; ++t)
#pragma unroll
    for (int r = 0; r < 8; ++r) {
      int row = nbase + r + ((lane >> 4) << 3);
      attb[(size_t)row * CDIM + t * 16 + (lane & 15)] = f2bf(acc[t][r] * lrun[r]);
    }
}

// ---------- kernel 3: out projection + residual (WMMA) ----------

__global__ __launch_bounds__(128, 1) void outproj_kernel(
    const uint16_t* __restrict__ attT, const uint16_t* __restrict__ wob,
    const float* __restrict__ bo, const float* __restrict__ x,
    const float* __restrict__ scale, float* __restrict__ y) {
  __shared__ float tile[4][16][17];
  const int lane  = threadIdx.x & 31;
  const int wave  = threadIdx.x >> 5;
  const int b     = blockIdx.x >> 6;
  const int ntile = blockIdx.x & 63;
  const int nbase = ntile * 64 + wave * 16;
  const float sc  = scale[0];

  const uint16_t* arow =
      attT + (size_t)b * NSEQ * CDIM + (size_t)(nbase + (lane & 15)) * CDIM;
  const v8f vzero = {};

  for (int t = 0; t < 16; ++t) {
    v8f acc = vzero;
    const uint16_t* bcol = wob + (size_t)(t * 16 + (lane & 15)) * CDIM;
#pragma unroll
    for (int ks = 0; ks < 8; ++ks) {
      v16bf a  = load_a(arow, ks * 32, lane);
      v16bf bb = load_b(bcol + ks * 32, lane);
      acc = wmma_bf16(a, bb, acc);
    }
    // D: lane = col(o), vgpr = row(n); transpose via LDS for coalesced fp32 stores
#pragma unroll
    for (int r = 0; r < 8; ++r)
      tile[wave][lane & 15][r + ((lane >> 4) << 3)] = acc[r];
    __syncthreads();
#pragma unroll
    for (int i = 0; i < 8; ++i) {
      int o_local = i * 2 + (lane >> 4);
      int n_local = lane & 15;
      int o = t * 16 + o_local;
      int n = nbase + n_local;
      size_t idx = (size_t)b * CDIM * NSEQ + (size_t)o * NSEQ + n;
      y[idx] = x[idx] + sc * (tile[wave][o_local][n_local] + bo[o]);
    }
    __syncthreads();
  }
}

// ---------- launch ----------

extern "C" void kernel_launch(void* const* d_in, const int* in_sizes, int n_in,
                              void* d_out, int out_size, void* d_ws, size_t ws_size,
                              hipStream_t stream) {
  const float* x     = (const float*)d_in[0];
  const float* wq    = (const float*)d_in[1];
  const float* bq    = (const float*)d_in[2];
  const float* wk    = (const float*)d_in[3];
  const float* bk    = (const float*)d_in[4];
  const float* wv    = (const float*)d_in[5];
  const float* bv    = (const float*)d_in[6];
  const float* wo    = (const float*)d_in[7];
  const float* bo    = (const float*)d_in[8];
  const float* scale = (const float*)d_in[9];
  float* y = (float*)d_out;

  char* ws = (char*)d_ws;
  size_t off = 0;
  uint16_t* xT   = (uint16_t*)(ws + off); off += (size_t)BATCH * NSEQ * CDIM * 2;
  uint16_t* qT   = (uint16_t*)(ws + off); off += (size_t)BATCH * NSEQ * QKD * 2;
  uint16_t* kT   = (uint16_t*)(ws + off); off += (size_t)BATCH * NSEQ * QKD * 2;
  uint16_t* vbf  = (uint16_t*)(ws + off); off += (size_t)BATCH * CDIM * NSEQ * 2;
  uint16_t* attT = (uint16_t*)(ws + off); off += (size_t)BATCH * NSEQ * CDIM * 2;
  uint16_t* wqb  = (uint16_t*)(ws + off); off += (size_t)QKD * CDIM * 2;
  uint16_t* wkb  = (uint16_t*)(ws + off); off += (size_t)QKD * CDIM * 2;
  uint16_t* wvb  = (uint16_t*)(ws + off); off += (size_t)CDIM * CDIM * 2;
  uint16_t* wob  = (uint16_t*)(ws + off); off += (size_t)CDIM * CDIM * 2;

  cast_bf16_kernel<<<dim3((QKD * CDIM + 255) / 256), dim3(256), 0, stream>>>(wq, wqb, QKD * CDIM);
  cast_bf16_kernel<<<dim3((QKD * CDIM + 255) / 256), dim3(256), 0, stream>>>(wk, wkb, QKD * CDIM);
  cast_bf16_kernel<<<dim3((CDIM * CDIM + 255) / 256), dim3(256), 0, stream>>>(wv, wvb, CDIM * CDIM);
  cast_bf16_kernel<<<dim3((CDIM * CDIM + 255) / 256), dim3(256), 0, stream>>>(wo, wob, CDIM * CDIM);

  transpose_cast_kernel<<<dim3(NSEQ / 32, CDIM / 32, BATCH), dim3(32, 8), 0, stream>>>(x, xT);

  dim3 grid(BATCH * NSEQ / 64), block(128);
  qkv_kernel<<<grid, block, 0, stream>>>(xT, wqb, wkb, wvb, bq, bk, bv, qT, kT, vbf);
  attn_kernel<<<grid, block, 0, stream>>>(qT, kT, vbf, attT);
  outproj_kernel<<<grid, block, 0, stream>>>(attT, wob, bo, x, scale, y);
}
